// Head_17884243821194
// MI455X (gfx1250) — matrix operand order repaired and checked by reading, not compiled, not run on hardware
//
#include <hip/hip_runtime.h>

// ---------------------------------------------------------------------------
// Single-head causal attention for MI455X (gfx1250, wave32, WMMA).
//   B=8, T=2048, C=1024, D=64.  fp32 in/out, bf16 WMMA compute, fp32 accum.
// Pipeline: prep_w  (w -> bf16 W^T; log2(e)*d_k^-0.5 folded into W_q so the
//                    whole softmax runs on raw v_exp_f32)
//           qkv_gemm (single pass over x computes Q,K,V together: 64MB read)
//           attn    (flash attention, 2-way key-split per query tile,
//                    log2-domain online softmax, LDS combine, WMMA matmuls)
// ---------------------------------------------------------------------------

typedef __attribute__((ext_vector_type(16))) __bf16 v16bf;
typedef __attribute__((ext_vector_type(8)))  __bf16 v8bf;
typedef __attribute__((ext_vector_type(8)))  float  v8f;

#define NEG_INF (-__builtin_inff())
// Raw hardware base-2 exponential (single v_exp_f32, no libm range guard).
// Softmax args are always <= 0; tiny results flushing to 0 is exactly right.
#define EXP2(x) __builtin_amdgcn_exp2f(x)

static constexpr int Bb = 8;
static constexpr int Tt = 2048;
static constexpr int Cc = 1024;
static constexpr int Dd = 64;
static constexpr int Mrows = Bb * Tt;            // 16384
// workspace layout (bf16 elements)
static constexpr size_t WT_ELEMS  = (size_t)3 * Dd * Cc;     // 196608
static constexpr size_t QKV_ELEMS = (size_t)Mrows * Dd;      // 1048576

// --------------------------------------------------------------------------
// Kernel 0: W[C][D] fp32 -> Wt[D][C] bf16 (x3). Scale log2(e)/sqrt(D) folded
// into W_q: scores come out as h*log2(e)/8, so softmax uses exp2 directly
// (one v_exp_f32, no v_mul by log2e). Mask-then-scale == scale-then-mask
// since -inf*s == -inf, and softmax ratios are unchanged in the log2 domain.
// --------------------------------------------------------------------------
__global__ __launch_bounds__(256)
void prep_w(const float* __restrict__ wq, const float* __restrict__ wk,
            const float* __restrict__ wv, __bf16* __restrict__ wt) {
  int idx = blockIdx.x * 256 + threadIdx.x;
  if (idx >= 3 * Cc * Dd) return;
  int mat = idx >> 16;           // 0..2
  int r   = idx & 65535;
  int k = r >> 6, n = r & 63;
  const float* w = (mat == 0) ? wq : (mat == 1) ? wk : wv;
  float v = w[k * Dd + n];
  if (mat == 0) v *= 0.18033688011112042f;   // log2(e) * d_k^-0.5 = log2(e)/8
  wt[(size_t)mat * Dd * Cc + (size_t)n * Cc + k] = (__bf16)v;
}

// --------------------------------------------------------------------------
// Kernel 1: fused QKV projection. One wave owns one 16-row tile of x and
// computes ALL THREE projections (12 accumulators, 12 WMMAs per K-chunk),
// so x is read exactly once from HBM (64 MB total, ~2.7us at 23.3 TB/s).
// Q,K stored [s][d] bf16 (LDS-staged coalesced stores); V stored [d][s].
// --------------------------------------------------------------------------
__global__ __launch_bounds__(256)
void qkv_gemm(const float* __restrict__ x, const __bf16* __restrict__ wt,
              __bf16* __restrict__ Qb, __bf16* __restrict__ Kb,
              __bf16* __restrict__ Vtb) {
  __shared__ __bf16 lds[8 * 16 * 64];            // 16 KB, 2 KB per wave
  const int wib  = threadIdx.x >> 5;
  const int lane = threadIdx.x & 31;
  const int wid  = blockIdx.x * 8 + wib;         // 0..1023 row tiles
  const int s0   = wid << 4;
  const int half = lane >> 4, lan16 = lane & 15;

  const float* xrow = x + (size_t)(s0 + lan16) * Cc;

  v8f acc[3][4] = {};
  for (int k0 = 0; k0 < Cc; k0 += 32) {
    // A fragment: 16x32 bf16 of x. Lane = row; elements are
    // K = k0+16*half .. +15 contiguous (matches B fragment K order).
    v16bf a;
    {
      const float* p0 = xrow + k0 + 16 * half;
      const float* p1 = p0 + 8;
#pragma unroll
      for (int i = 0; i < 8; ++i) {
        a[i]     = (__bf16)p0[i];
        a[i + 8] = (__bf16)p1[i];
      }
    }
#pragma unroll
    for (int mat = 0; mat < 3; ++mat) {
#pragma unroll
      for (int sub = 0; sub < 4; ++sub) {
        // B fragment: 32x16 bf16 of W^T. Lane = col n; elements = K
        // (16 contiguous bf16 thanks to the [n][k] transposed layout).
        const __bf16* bp = wt + (size_t)mat * Dd * Cc +
                           (size_t)(sub * 16 + lan16) * Cc + k0 + 16 * half;
        v16bf bfr = *(const v16bf*)bp;
        acc[mat][sub] = __builtin_amdgcn_wmma_f32_16x16x32_bf16(
            false, a, false, bfr, (short)0, acc[mat][sub], false, false);
      }
    }
  }

  // ---- Q then K: stage C fragments in LDS, coalesced 16B stores [s][d] ----
  __bf16* lt = lds + wib * 16 * 64;
#pragma unroll
  for (int mat = 0; mat < 2; ++mat) {
#pragma unroll
    for (int sub = 0; sub < 4; ++sub)
#pragma unroll
      for (int r = 0; r < 8; ++r)
        lt[(r + 8 * half) * 64 + sub * 16 + lan16] = (__bf16)acc[mat][sub][r];
    asm volatile("s_wait_dscnt 0" ::: "memory");
    __bf16* dst = (mat == 0 ? Qb : Kb) + (size_t)s0 * Dd;
#pragma unroll
    for (int j = 0; j < 4; ++j) {
      int chunk = lane + 32 * j;                 // 0..127
      int row = chunk >> 3, cc8 = chunk & 7;
      *(v8bf*)(dst + row * 64 + cc8 * 8) = *(const v8bf*)(lt + row * 64 + cc8 * 8);
    }
    asm volatile("" ::: "memory");  // keep next mat's DS stores after loads
  }

  // ---- V stored transposed [b][d][s]: C-frag rows are contiguous in s ----
  {
    int b  = s0 >> 11;
    int sl = s0 & 2047;
#pragma unroll
    for (int sub = 0; sub < 4; ++sub) {
      int d = sub * 16 + lan16;
      v8bf vv;
#pragma unroll
      for (int r = 0; r < 8; ++r) vv[r] = (__bf16)acc[2][sub][r];
      *(v8bf*)(Vtb + ((size_t)b * Dd + d) * Tt + sl + 8 * half) = vv;
    }
  }
}

// --------------------------------------------------------------------------
// Kernel 2: flash attention, 2-way key split, log2-domain online softmax.
// Each query tile (16 rows) is handled by a PAIR of waves: sub=0 takes even
// 32-key tiles, sub=1 odd (causal work balanced to +-1 tile). Each keeps
// partial (o, m, l); partials merged via the flash combine through LDS.
// S = Q K^T (2 chained WMMAs x 2 subtiles), P transposed through LDS,
// O += P V (4 WMMAs). 2048 waves total. Every exponential is a single
// v_exp_f32: scores pre-scaled by log2(e)/8 in W_q, raw hw exp2 builtin.
// --------------------------------------------------------------------------
__global__ __launch_bounds__(256)
void attn(const __bf16* __restrict__ Qb, const __bf16* __restrict__ Kb,
          const __bf16* __restrict__ Vtb, float* __restrict__ out) {
  __shared__ __bf16 plds[8 * 16 * 32];           // 8 KB, 1 KB per wave
  __shared__ float  comb[4 * 32 * 48];           // 24 KB, pair exchange
  const int wib  = threadIdx.x >> 5;
  const int lane = threadIdx.x & 31;
  const int gw   = blockIdx.x * 8 + wib;         // 0..2047
  const int pair = gw >> 1;                      // query tile id 0..1023
  const int sub  = gw & 1;                       // key-range split
  const int b    = pair >> 7;
  const int q0   = (pair & 127) << 4;
  const int half = lane >> 4, lan16 = lane & 15;

  const __bf16* Q  = Qb  + (size_t)b * Tt * Dd;
  const __bf16* K  = Kb  + (size_t)b * Tt * Dd;
  const __bf16* Vt = Vtb + (size_t)b * Dd * Tt;
  __bf16* pl = plds + wib * 16 * 32;
  float*  cb = comb + (wib >> 1) * (32 * 48) + lane * 48;

  // Q A-fragments, reduction dim 64 = two chunks of 32 (scale pre-folded).
  v16bf aq[2];
  {
    const __bf16* qrow = Q + (size_t)(q0 + lan16) * Dd;
#pragma unroll
    for (int kc = 0; kc < 2; ++kc) {
      v8bf lo = *(const v8bf*)(qrow + 32 * kc + 16 * half);
      v8bf hi = *(const v8bf*)(qrow + 32 * kc + 16 * half + 8);
#pragma unroll
      for (int i = 0; i < 8; ++i) { aq[kc][i] = lo[i]; aq[kc][i + 8] = hi[i]; }
    }
  }

  v8f o[4] = {};
  float m[8], l[8];
#pragma unroll
  for (int r = 0; r < 8; ++r) { m[r] = NEG_INF; l[r] = 0.0f; }

  const int ktmax = (q0 + 15) >> 5;
  for (int kt = sub; kt <= ktmax; kt += 2) {
    const int n0 = kt << 5;
    // --- S = Q K^T  (16x32 log2-scaled scores, two 16x16 subtiles) ---
    v8f s[2];
#pragma unroll
    for (int t = 0; t < 2; ++t) {
      const __bf16* krow = K + (size_t)(n0 + 16 * t + lan16) * Dd;
      v16bf b0 = *(const v16bf*)(krow + 16 * half);
      v16bf b1 = *(const v16bf*)(krow + 32 + 16 * half);
      v8f z = {};
      z    = __builtin_amdgcn_wmma_f32_16x16x32_bf16(false, aq[0], false, b0,
                                                     (short)0, z, false, false);
      s[t] = __builtin_amdgcn_wmma_f32_16x16x32_bf16(false, aq[1], false, b1,
                                                     (short)0, z, false, false);
    }
    // --- causal mask (only boundary tiles need it) ---
    if (n0 + 31 > q0) {
#pragma unroll
      for (int t = 0; t < 2; ++t) {
        int ki = n0 + 16 * t + lan16;
#pragma unroll
        for (int r = 0; r < 8; ++r)
          if (ki > q0 + r + 8 * half) s[t][r] = NEG_INF;
      }
    }
    // --- log2-domain online softmax: row max, alpha, p, row sum ---
    float alpha[8];
#pragma unroll
    for (int r = 0; r < 8; ++r) {
      float v = fmaxf(s[0][r], s[1][r]);
#pragma unroll
      for (int off = 1; off < 16; off <<= 1)
        v = fmaxf(v, __shfl_xor(v, off, 32));
      float mn = fmaxf(m[r], v);
      alpha[r] = EXP2(m[r] - mn);                // single v_exp_f32
      m[r] = mn;
    }
    float rs[8];
#pragma unroll
    for (int r = 0; r < 8; ++r) rs[r] = 0.0f;
#pragma unroll
    for (int t = 0; t < 2; ++t)
#pragma unroll
      for (int r = 0; r < 8; ++r) {
        float p = EXP2(s[t][r] - m[r]);          // single v_exp_f32
        rs[r] += p;
        pl[(r + 8 * half) * 32 + 16 * t + lan16] = (__bf16)p;   // C->A transpose
      }
#pragma unroll
    for (int r = 0; r < 8; ++r) {
      float v = rs[r];
#pragma unroll
      for (int off = 1; off < 16; off <<= 1) v += __shfl_xor(v, off, 32);
      l[r] = l[r] * alpha[r] + v;
    }
#pragma unroll
    for (int dt = 0; dt < 4; ++dt)
#pragma unroll
      for (int r = 0; r < 8; ++r) o[dt][r] *= alpha[r];
    asm volatile("s_wait_dscnt 0" ::: "memory");
    // --- P as A-fragment (16x32), V^T as B-fragments, O += P V ---
    v16bf ap;
    {
      const __bf16* pr = pl + lan16 * 32 + 16 * half;
      v8bf lo = *(const v8bf*)pr;
      v8bf hi = *(const v8bf*)(pr + 8);
#pragma unroll
      for (int i = 0; i < 8; ++i) { ap[i] = lo[i]; ap[i + 8] = hi[i]; }
    }
#pragma unroll
    for (int dt = 0; dt < 4; ++dt) {
      const __bf16* vp = Vt + (size_t)(dt * 16 + lan16) * Tt + n0 + 16 * half;
      v16bf bv = *(const v16bf*)vp;
      o[dt] = __builtin_amdgcn_wmma_f32_16x16x32_bf16(false, ap, false, bv,
                                                      (short)0, o[dt], false, false);
    }
  }

  // --- pair combine: odd wave publishes partial (o,m,l), even wave merges ---
  if (sub) {
#pragma unroll
    for (int dt = 0; dt < 4; ++dt)
#pragma unroll
      for (int r = 0; r < 8; ++r) cb[dt * 8 + r] = o[dt][r];
#pragma unroll
    for (int r = 0; r < 8; ++r) { cb[32 + r] = m[r]; cb[40 + r] = l[r]; }
  }
  __syncthreads();
  if (!sub) {
    float* ob = out + ((size_t)b * Tt + q0) * Dd;
#pragma unroll
    for (int r = 0; r < 8; ++r) {
      float m2 = cb[32 + r];
      float mm = fmaxf(m[r], m2);
      float a1 = EXP2(m[r] - mm);
      float a2 = EXP2(m2 - mm);
      float ll = l[r] * a1 + cb[40 + r] * a2;
      float inv = 1.0f / ll;
#pragma unroll
      for (int dt = 0; dt < 4; ++dt)
        ob[(r + 8 * half) * Dd + dt * 16 + lan16] =
            (o[dt][r] * a1 + cb[dt * 8 + r] * a2) * inv;
    }
  }
}

// --------------------------------------------------------------------------
extern "C" void kernel_launch(void* const* d_in, const int* in_sizes, int n_in,
                              void* d_out, int out_size, void* d_ws, size_t ws_size,
                              hipStream_t stream) {
  const float* x  = (const float*)d_in[0];
  const float* wq = (const float*)d_in[1];
  const float* wk = (const float*)d_in[2];
  const float* wv = (const float*)d_in[3];
  float* out = (float*)d_out;

  __bf16* wt  = (__bf16*)d_ws;                 // [3][64][1024]
  __bf16* Qb  = wt + WT_ELEMS;                 // [16384][64]
  __bf16* Kb  = Qb + QKV_ELEMS;                // [16384][64]
  __bf16* Vtb = Kb + QKV_ELEMS;                // [8][64][2048]

  prep_w<<<(3 * Cc * Dd + 255) / 256, 256, 0, stream>>>(wq, wk, wv, wt);
  qkv_gemm<<<(Mrows / 16) / 8, 256, 0, stream>>>(x, wt, Qb, Kb, Vtb);
  attn<<<(2 * Bb * (Tt / 16)) / 8, 256, 0, stream>>>(Qb, Kb, Vtb, out);
}
